// LinearAttention_46583215292776
// MI455X (gfx1250) — matrix operand order repaired and checked by reading, not compiled, not run on hardware
//
#include <hip/hip_runtime.h>
#include <hip/hip_bf16.h>

typedef __attribute__((ext_vector_type(16))) __bf16 v16bf;
typedef __attribute__((ext_vector_type(8)))  __bf16 v8bf;
typedef __attribute__((ext_vector_type(8)))  float  v8f;
typedef __attribute__((ext_vector_type(4)))  unsigned int v4u;
typedef __attribute__((ext_vector_type(4)))  int v4i;

constexpr int Bc   = 16;
constexpr int Hh   = 64;
constexpr int Ww   = 64;
constexpr int Cc   = 256;
constexpr int Nn   = Hh * Ww;          // 4096
constexpr int NH   = 8;
constexpr int Mrows = Bc * Nn;         // 65536
constexpr int Ncols = 2 * Cc;          // 512
constexpr int Kdim  = Cc;              // 256

// --- gfx1250 async global->LDS path (guarded; falls back to plain copies) ---
#if __has_builtin(__builtin_amdgcn_global_load_async_to_lds_b128) && \
    __has_builtin(__builtin_amdgcn_s_wait_asynccnt)
#define USE_ASYNC_LDS 1
__device__ __forceinline__ void g2l_b128(const void* g, void* l) {
    __builtin_amdgcn_global_load_async_to_lds_b128(
        (__attribute__((address_space(1))) v4i*)g,
        (__attribute__((address_space(3))) v4i*)l, 0, 0);
}
#else
#define USE_ASYNC_LDS 0
#endif

__device__ __forceinline__ unsigned pack2bf(float a, float b) {
    unsigned short lo = __builtin_bit_cast(unsigned short, (__bf16)a);
    unsigned short hi = __builtin_bit_cast(unsigned short, (__bf16)b);
    return (unsigned)lo | ((unsigned)hi << 16);
}

// ---------------------------------------------------------------------------
// Pack fp32 -> bf16, 2 elements/thread, packed 4B stores.
// ---------------------------------------------------------------------------
__global__ void pack_bf16_kernel(const float* __restrict__ in,
                                 __bf16* __restrict__ out) {
    const int idx = blockIdx.x * blockDim.x + threadIdx.x;   // pair index
    const float2 v = ((const float2*)in)[idx];
    ((unsigned*)out)[idx] = pack2bf(v.x, v.y);
}

// ---------------------------------------------------------------------------
// Kernel A: qk = elu(x @ Wqk^T + bqk) + 1   (bf16 WMMA, fp32 accumulate)
// 8 waves/block, wave = 16(M) x 64(N), block = 128(M) x 64(N).
// B tile staged through LDS (shared by all waves).
// ---------------------------------------------------------------------------
__global__ void qk_gemm_kernel(const __bf16* __restrict__ xbf,
                               const __bf16* __restrict__ Wbf,
                               const float* __restrict__ bqk,
                               float* __restrict__ qk) {
    __shared__ __bf16 Bs[64 * 32];
    const int tid    = threadIdx.x;
    const int lane   = tid & 31;
    const int wave   = tid >> 5;
    const int laneLo = lane & 15;
    const int hiA    = (lane >= 16) ? 8 : 0;
    const int hiB    = (lane >= 16) ? 16 : 0;
    const int m0     = blockIdx.x * 128 + wave * 16;
    const int n0     = blockIdx.y * 64;
    const int arow   = m0 + laneLo;
    const int scol   = tid >> 2;         // 0..63 staged column
    const int sseg   = (tid & 3) * 8;    // 0,8,16,24 K-segment

    v8f acc[4] = {};

    for (int kk = 0; kk < Kdim; kk += 32) {
        const __bf16* wsrc = Wbf + (size_t)(n0 + scol) * Kdim + kk + sseg;
#if USE_ASYNC_LDS
        g2l_b128(wsrc, &Bs[scol * 32 + sseg]);
        __builtin_amdgcn_s_wait_asynccnt(0);
#else
        *(v4u*)&Bs[scol * 32 + sseg] = *(const v4u*)wsrc;
#endif
        __syncthreads();

        const __bf16* ap = xbf + (size_t)arow * Kdim + kk;
        __builtin_prefetch(ap + 32, 0, 1);
        const v8bf a0 = *(const v8bf*)(ap + 2 * hiA);
        const v8bf a1 = *(const v8bf*)(ap + 8 + 2 * hiA);
        const v16bf a = __builtin_shufflevector(a0, a1,
            0, 1, 2, 3, 4, 5, 6, 7, 8, 9, 10, 11, 12, 13, 14, 15);

        #pragma unroll
        for (int t = 0; t < 4; ++t) {
            const v16bf bm = *(const v16bf*)(&Bs[(t * 16 + laneLo) * 32 + hiB]);
            acc[t] = __builtin_amdgcn_wmma_f32_16x16x32_bf16(
                false, a, false, bm, (short)0, acc[t], false, false);
        }
        __syncthreads();
    }

    #pragma unroll
    for (int t = 0; t < 4; ++t) {
        const int col = n0 + t * 16 + laneLo;
        const float bias = bqk[col];
        #pragma unroll
        for (int r = 0; r < 8; ++r) {
            const int row = m0 + r + hiA;
            float v = acc[t][r] + bias;
            v = (v > 0.f) ? (v + 1.f) : __expf(v);   // elu(v)+1
            qk[(size_t)row * Ncols + col] = v;
        }
    }
}

// ---------------------------------------------------------------------------
// Kernel B: RoPE on q,k -> bf16 (packed 4B stores).
// grid = 65536 blocks (b*n), 128 threads (channel pairs)
// ---------------------------------------------------------------------------
__global__ void rope_pack_kernel(const float* __restrict__ qk,
                                 __bf16* __restrict__ qrope,
                                 __bf16* __restrict__ krope) {
    const int row = blockIdx.x;            // b*n
    const int p   = row & (Nn - 1);
    const int i   = p >> 6;
    const int j   = p & 63;
    const int t   = threadIdx.x;           // pair index 0..127
    const int kidx = t & 63;

    const float theta = __powf(10000.f, -(float)kidx * (1.f / 64.f));
    const float ang   = ((t < 64) ? (float)i : (float)j) * theta;
    float sn, cs;
    __sincosf(ang, &sn, &cs);

    const float2* qrow = (const float2*)(qk + (size_t)row * Ncols);
    const float2 q2 = qrow[t];
    ((unsigned*)(qrope + (size_t)row * Cc))[t] =
        pack2bf(q2.x * cs - q2.y * sn, q2.x * sn + q2.y * cs);
    const float2 k2 = qrow[128 + t];
    ((unsigned*)(krope + (size_t)row * Cc))[t] =
        pack2bf(k2.x * cs - k2.y * sn, k2.x * sn + k2.y * cs);
}

// ---------------------------------------------------------------------------
// Kernel C: k_mean[b, ch] = mean_n k[b, n, ch].  16 blocks, coalesced in ch.
// ---------------------------------------------------------------------------
__global__ void kmean_kernel(const float* __restrict__ qk,
                             float* __restrict__ kmean) {
    const int b = blockIdx.x;
    const int ch = threadIdx.x;
    const float* base = qk + ((size_t)b * Nn) * Ncols + Cc + ch;
    float s = 0.f;
    for (int p = 0; p < Nn; ++p) s += base[(size_t)p * Ncols];
    kmean[b * Cc + ch] = s * (1.f / Nn);
}

// ---------------------------------------------------------------------------
// Kernel Z: z[b,h,n] = 1 / (q[b,n,h*32:+32] . kmean[b,h*32:+32] + 1e-6)
// ---------------------------------------------------------------------------
__global__ void z_kernel(const float* __restrict__ qk,
                         const float* __restrict__ kmean,
                         float* __restrict__ z) {
    const int idx = blockIdx.x * blockDim.x + threadIdx.x;   // b*h*n
    const int p  = idx & (Nn - 1);
    const int bh = idx >> 12;
    const int h  = bh & 7, b = bh >> 3;
    const float* qrow = qk + ((size_t)(b * Nn + p)) * Ncols + h * 32;
    const float* km   = kmean + b * Cc + h * 32;
    float s = 0.f;
    #pragma unroll
    for (int d = 0; d < 32; ++d) s += qrow[d] * km[d];
    z[idx] = 1.f / (s + 1e-6f);
}

// ---------------------------------------------------------------------------
// Kernel D: kv[b,h] = (1/n) * k_rope^T (32 x n) @ v (n x 32)  via WMMA.
// 128 blocks (b*h); 8 waves each own an n-chunk of 512.
// Tiles staged coalesced through per-wave LDS slabs; result stored
// TRANSPOSED in bf16 (kvt[e][d]) so the out-kernel B load is contiguous.
// ---------------------------------------------------------------------------
__global__ void kv_kernel(const __bf16* __restrict__ krope,
                          const __bf16* __restrict__ xbf,
                          __bf16* __restrict__ kvt) {
    __shared__ __bf16 kt[8][1024];
    __shared__ __bf16 vt[8][1024];
    __shared__ float  red[8][1024];
    const int bh = blockIdx.x;
    const int h = bh & 7, b = bh >> 3;
    const int lane = threadIdx.x & 31, wave = threadIdx.x >> 5;
    const int laneLo = lane & 15;
    const int hiA = (lane >= 16) ? 8 : 0;
    const int hiB = (lane >= 16) ? 16 : 0;

    const __bf16* kbase = krope + ((size_t)b * Nn) * Cc + h * 32;
    const __bf16* vbase = xbf   + ((size_t)b * Nn) * Cc + h * 32;

    v8f acc[2][2] = {};
    const int nBeg = wave * 512, nEnd = nBeg + 512;
    for (int n0 = nBeg; n0 < nEnd; n0 += 32) {
        // lane stages row (n0+lane): 32 bf16 = 64B, coalesced
        const __bf16* gk = kbase + (size_t)(n0 + lane) * Cc;
        const __bf16* gv = vbase + (size_t)(n0 + lane) * Cc;
#if USE_ASYNC_LDS
        #pragma unroll
        for (int s = 0; s < 4; ++s) {
            g2l_b128(gk + s * 8, &kt[wave][lane * 32 + s * 8]);
            g2l_b128(gv + s * 8, &vt[wave][lane * 32 + s * 8]);
        }
        __builtin_amdgcn_s_wait_asynccnt(0);
#else
        {
            v4u* dk = (v4u*)&kt[wave][lane * 32];
            v4u* dv = (v4u*)&vt[wave][lane * 32];
            const v4u* sk = (const v4u*)gk;
            const v4u* sv = (const v4u*)gv;
            #pragma unroll
            for (int s = 0; s < 4; ++s) { dk[s] = sk[s]; dv[s] = sv[s]; }
        }
#endif
        v16bf a[2], bm[2];
        #pragma unroll
        for (int ti = 0; ti < 2; ++ti) {
            const int m = ti * 16 + laneLo;   // d index
            #pragma unroll
            for (int e = 0; e < 16; ++e) {
                const int kk = hiB + e;
                a[ti][e] = kt[wave][kk * 32 + m];
            }
        }
        #pragma unroll
        for (int tj = 0; tj < 2; ++tj) {
            const int col = tj * 16 + laneLo; // e index
            #pragma unroll
            for (int e = 0; e < 16; ++e)
                bm[tj][e] = vt[wave][(hiB + e) * 32 + col];
        }
        #pragma unroll
        for (int ti = 0; ti < 2; ++ti)
            #pragma unroll
            for (int tj = 0; tj < 2; ++tj)
                acc[ti][tj] = __builtin_amdgcn_wmma_f32_16x16x32_bf16(
                    false, a[ti], false, bm[tj], (short)0, acc[ti][tj], false, false);
    }

    #pragma unroll
    for (int ti = 0; ti < 2; ++ti)
        #pragma unroll
        for (int tj = 0; tj < 2; ++tj)
            #pragma unroll
            for (int r = 0; r < 8; ++r) {
                const int m = ti * 16 + r + hiA;
                const int e = tj * 16 + laneLo;
                red[wave][m * 32 + e] = acc[ti][tj][r];
            }
    __syncthreads();
    for (int q = threadIdx.x; q < 1024; q += 256) {
        float s = 0.f;
        #pragma unroll
        for (int w = 0; w < 8; ++w) s += red[w][q];
        const int m = q >> 5, e = q & 31;
        kvt[(size_t)bh * 1024 + e * 32 + m] = (__bf16)(s * (1.f / Nn));
    }
}

// ---------------------------------------------------------------------------
// Kernel F: lepe depthwise 3x3 conv of v (=x) -> d_out (base value)
// ---------------------------------------------------------------------------
__global__ void lepe_kernel(const float* __restrict__ x,
                            const float* __restrict__ w,
                            const float* __restrict__ lb,
                            float* __restrict__ out) {
    const int row = blockIdx.x;           // b*n
    const int b = row >> 12;
    const int p = row & (Nn - 1);
    const int i = p >> 6, j = p & 63;
    const int ch = threadIdx.x;
    const float* wc = w + ch * 9;
    float s = lb[ch];
    #pragma unroll
    for (int di = -1; di <= 1; ++di) {
        const int ii = i + di;
        if (ii < 0 || ii >= Hh) continue;
        #pragma unroll
        for (int dj = -1; dj <= 1; ++dj) {
            const int jj = j + dj;
            if (jj < 0 || jj >= Ww) continue;
            s += x[((size_t)(b * Nn + ii * Ww + jj)) * Cc + ch] * wc[(di + 1) * 3 + (dj + 1)];
        }
    }
    out[(size_t)row * Cc + ch] = s;
}

// ---------------------------------------------------------------------------
// Kernel E: out[b,n,h*32+e] += z[b,h,n] * (q_rope (n x 32) @ kv (32 x 32))
// kvt is bf16, transposed: B operand is one contiguous 32B load.
// ---------------------------------------------------------------------------
__global__ void out_kernel(const __bf16* __restrict__ qrope,
                           const __bf16* __restrict__ kvt,
                           const float* __restrict__ z,
                           float* __restrict__ out) {
    const int bh = blockIdx.x;
    const int h = bh & 7, b = bh >> 3;
    const int nblk = blockIdx.y;
    const int lane = threadIdx.x & 31, wave = threadIdx.x >> 5;
    const int laneLo = lane & 15;
    const int hiA = (lane >= 16) ? 8 : 0;
    const int hiB = (lane >= 16) ? 16 : 0;

    const int rowA = nblk * 128 + wave * 16 + laneLo;
    const __bf16* qbase = qrope + ((size_t)(b * Nn + rowA)) * Cc + h * 32;
    const v8bf a0 = *(const v8bf*)(qbase + 2 * hiA);
    const v8bf a1 = *(const v8bf*)(qbase + 8 + 2 * hiA);
    const v16bf a = __builtin_shufflevector(a0, a1,
        0, 1, 2, 3, 4, 5, 6, 7, 8, 9, 10, 11, 12, 13, 14, 15);

    const __bf16* kvb = kvt + (size_t)bh * 1024;
    v8f acc[2] = {};
    #pragma unroll
    for (int tj = 0; tj < 2; ++tj) {
        const int col = tj * 16 + laneLo;
        const v16bf bm = *(const v16bf*)(kvb + col * 32 + hiB);
        acc[tj] = __builtin_amdgcn_wmma_f32_16x16x32_bf16(
            false, a, false, bm, (short)0, acc[tj], false, false);
    }

    #pragma unroll
    for (int tj = 0; tj < 2; ++tj) {
        const int col = h * 32 + tj * 16 + laneLo;
        #pragma unroll
        for (int r = 0; r < 8; ++r) {
            const int rowm = nblk * 128 + wave * 16 + r + hiA;
            const float zz = z[(size_t)bh * Nn + rowm];
            const size_t oi = ((size_t)(b * Nn + rowm)) * Cc + col;
            out[oi] += acc[tj][r] * zz;
        }
    }
}

// ---------------------------------------------------------------------------
extern "C" void kernel_launch(void* const* d_in, const int* in_sizes, int n_in,
                              void* d_out, int out_size, void* d_ws, size_t ws_size,
                              hipStream_t stream) {
    const float* x   = (const float*)d_in[0];
    const float* Wqk = (const float*)d_in[1];
    const float* bqk = (const float*)d_in[2];
    const float* lw  = (const float*)d_in[3];
    const float* lb  = (const float*)d_in[4];
    // d_in[5], d_in[6] are H, W (fixed at 64 per setup_inputs)

    char* ws = (char*)d_ws;
    size_t off = 0;
    float*  qk    = (float*)(ws + off);  off += (size_t)Mrows * Ncols * sizeof(float);   // 128 MB
    __bf16* xbf   = (__bf16*)(ws + off); off += (size_t)Mrows * Cc * sizeof(__bf16);     // 32 MB
    __bf16* Wbf   = (__bf16*)(ws + off); off += (size_t)Ncols * Kdim * sizeof(__bf16);   // 256 KB
    __bf16* qrope = (__bf16*)(ws + off); off += (size_t)Mrows * Cc * sizeof(__bf16);     // 32 MB
    __bf16* krope = (__bf16*)(ws + off); off += (size_t)Mrows * Cc * sizeof(__bf16);     // 32 MB
    float*  kmean = (float*)(ws + off);  off += (size_t)Bc * Cc * sizeof(float);
    __bf16* kvt   = (__bf16*)(ws + off); off += (size_t)Bc * NH * 32 * 32 * sizeof(__bf16);
    float*  zw    = (float*)(ws + off);  off += (size_t)Bc * NH * Nn * sizeof(float);    // 2 MB

    float* out = (float*)d_out;

    pack_bf16_kernel<<<((size_t)Mrows * Cc / 2) / 256, 256, 0, stream>>>(x, xbf);
    pack_bf16_kernel<<<((size_t)Ncols * Kdim / 2) / 256, 256, 0, stream>>>(Wqk, Wbf);
    qk_gemm_kernel  <<<dim3(Mrows / 128, Ncols / 64), 256, 0, stream>>>(xbf, Wbf, bqk, qk);
    rope_pack_kernel<<<Mrows, 128, 0, stream>>>(qk, qrope, krope);
    kmean_kernel    <<<Bc, 256, 0, stream>>>(qk, kmean);
    z_kernel        <<<(Bc * NH * Nn) / 256, 256, 0, stream>>>(qk, kmean, zw);
    kv_kernel       <<<Bc * NH, 256, 0, stream>>>(krope, xbf, kvt);
    lepe_kernel     <<<Mrows, 256, 0, stream>>>(x, lw, lb, out);
    out_kernel      <<<dim3(Bc * NH, Nn / 128), 256, 0, stream>>>(qrope, kvt, zw, out);
}
